// Attention_60979945668745
// MI455X (gfx1250) — matrix operand order repaired and checked by reading, not compiled, not run on hardware
//
#include <hip/hip_runtime.h>

// ---------------------------------------------------------------------------
// MI455X (gfx1250) attention block:
//   q/k/v = x@W^T (bf16 WMMA) + RoPE  ->  flash attention (bf16 WMMA, fp32
//   online softmax)  ->  out = attn@wo^T (bf16 WMMA, fp32 out)
// All matmuls via v_wmma_f32_16x16x32_bf16 with fp32 accumulation.
// GEMM wave tile is 32x64 (two M-subtiles share every B tile):
//   12 x global_load_b128 per 8 WMMAs = 1.5 loads/WMMA.
// ---------------------------------------------------------------------------

typedef __attribute__((ext_vector_type(16))) __bf16 v16bf;
typedef __attribute__((ext_vector_type(8)))  float  v8f;
typedef __attribute__((ext_vector_type(4)))  unsigned int v4u;

constexpr int DIM   = 2048;
constexpr int NH    = 32;
constexpr int HD    = 64;
constexpr int BATCH = 2;
constexpr int SEQ   = 2048;
constexpr int TOK   = BATCH * SEQ;          // 4096 tokens
constexpr float SM_SCALE = 0.125f;          // 1/sqrt(64)
constexpr float NEG_INF  = -1e9f;

// Load 16 bf16 (two 16B chunks) into a WMMA operand register.
__device__ __forceinline__ v16bf load_tile16(const __bf16* p, int off_a, int off_b) {
  union { v4u u[2]; v16bf v; } r;
  r.u[0] = *reinterpret_cast<const v4u*>(p + off_a);   // global_load_b128 / ds_load_b128
  r.u[1] = *reinterpret_cast<const v4u*>(p + off_b);
  return r.v;
}

__device__ __forceinline__ v8f wmma_bf16(v16bf a, v16bf b, v8f c) {
  return __builtin_amdgcn_wmma_f32_16x16x32_bf16(
      /*neg_a=*/false, a, /*neg_b=*/false, b,
      /*c_mod=*/(short)0, c, /*reuse_a=*/false, /*reuse_b=*/false);
}

// ---------------------------------------------------------------------------
// fp32 -> bf16 cast, 4 elements/thread (b128 load, b64 store)
// ---------------------------------------------------------------------------
__global__ __launch_bounds__(256)
void cvt_f32_bf16_k(const float* __restrict__ src, __bf16* __restrict__ dst, int n4) {
  int i = blockIdx.x * 256 + threadIdx.x;
  if (i >= n4) return;
  float4 v = reinterpret_cast<const float4*>(src)[i];
  union { __bf16 h[4]; unsigned long long u; } r;
  r.h[0] = (__bf16)v.x; r.h[1] = (__bf16)v.y;
  r.h[2] = (__bf16)v.z; r.h[3] = (__bf16)v.w;
  *reinterpret_cast<unsigned long long*>(dst + 4 * (size_t)i) = r.u;
}

// ---------------------------------------------------------------------------
// Y(4096x2048) = A(4096x2048,bf16) @ W^T(2048x2048 bf16, row-major (n,k)).
// Wave tile: 32 rows x 64 cols (one head wide). Block = 8 waves = 256x64.
// MODE 0: RoPE epilogue -> bf16 (b,h,s,hd)        [Q and K]
// MODE 1: transpose epilogue -> bf16 (b,h,hd,s)   [V]
// MODE 2: fp32 store (token, dim)                 [final projection]
// ---------------------------------------------------------------------------
template <int MODE>
__global__ __launch_bounds__(256)
void gemm_bf16_k(const __bf16* __restrict__ A, const __bf16* __restrict__ W,
                 void* __restrict__ outp, const int* __restrict__ startp) {
  const int lane = threadIdx.x & 31;
  const int wave = threadIdx.x >> 5;
  const int half = lane >> 4;       // which 16-lane group
  const int col  = lane & 15;       // A row / B col / C col within tile
  const int n0   = blockIdx.x * 64;
  const int m0   = blockIdx.y * 256 + wave * 32;

  // A-operand per-lane chunk offsets (ISA 16-bit A 16x32 layout)
  const int oa1 = half * 8, oa2 = 16 + half * 8;
  // B-operand per-lane chunk offsets (same K order as A)
  const int ob1 = half * 8, ob2 = 16 + half * 8;

  const __bf16* arow0 = A + (size_t)(m0 + col) * DIM;
  const __bf16* arow1 = arow0 + (size_t)16 * DIM;
  const __bf16* wrow  = W + (size_t)(n0 + col) * DIM;

  v8f acc[2][4];
  const v8f vzero = {0.f, 0.f, 0.f, 0.f, 0.f, 0.f, 0.f, 0.f};
#pragma unroll
  for (int mt = 0; mt < 2; ++mt)
#pragma unroll
    for (int t = 0; t < 4; ++t) acc[mt][t] = vzero;

  for (int kc = 0; kc < DIM; kc += 32) {
    __builtin_prefetch(arow0 + kc + 64, 0, 1);          // global_prefetch_b8
    __builtin_prefetch(wrow  + kc + 64, 0, 1);
    v16bf a0 = load_tile16(arow0, kc + oa1, kc + oa2);
    v16bf a1 = load_tile16(arow1, kc + oa1, kc + oa2);
#pragma unroll
    for (int t = 0; t < 4; ++t) {
      v16bf b = load_tile16(wrow + (size_t)t * 16 * DIM, kc + ob1, kc + ob2);
      acc[0][t] = wmma_bf16(a0, b, acc[0][t]);
      acc[1][t] = wmma_bf16(a1, b, acc[1][t]);
    }
  }

  // C layout: vgpr v -> row (v + half*8), lane%16 -> col.
#pragma unroll
  for (int mt = 0; mt < 2; ++mt) {
    const int mb = m0 + mt * 16;
    if constexpr (MODE == 0) {               // RoPE + (b,h,s,hd) bf16 store
      __bf16* qout = (__bf16*)outp;
      const int sp = startp[0];
      const int h  = n0 >> 6;                // 64 cols == one head
      // reference broadcast quirk: one angle per (token, head):
      // theta = (start_pos + s) * 10000^(-h/32)
      const float fb = __expf(-(float)h * (9.2103403719761836f / 32.0f));
      float cv[8], sv[8];
      int oidx[8];
#pragma unroll
      for (int v = 0; v < 8; ++v) {
        int token = mb + v + half * 8;
        int b = token / SEQ, s = token - b * SEQ;
        float th = (float)(sp + s) * fb;
        __sincosf(th, &sv[v], &cv[v]);
        oidx[v] = ((b * NH + h) * SEQ + s) * HD;
      }
      const float sgn = (lane & 1) ? 1.0f : -1.0f;  // even col: -sin, odd: +sin
#pragma unroll
      for (int t = 0; t < 4; ++t)
#pragma unroll
        for (int v = 0; v < 8; ++v) {
          float val = acc[mt][t][v];
          float par = __shfl_xor(val, 1);           // even/odd rotation pair
          float r = val * cv[v] + sgn * par * sv[v];
          qout[oidx[v] + t * 16 + col] = (__bf16)r;
        }
    } else if constexpr (MODE == 1) {        // V transposed (b,h,hd,s)
      __bf16* vout = (__bf16*)outp;
      const int h = n0 >> 6;
      int oidx[8];
#pragma unroll
      for (int v = 0; v < 8; ++v) {
        int token = mb + v + half * 8;
        int b = token / SEQ, s = token - b * SEQ;
        oidx[v] = (b * NH + h) * HD * SEQ + s;
      }
#pragma unroll
      for (int t = 0; t < 4; ++t)
#pragma unroll
        for (int v = 0; v < 8; ++v)
          vout[oidx[v] + (t * 16 + col) * SEQ] = (__bf16)acc[mt][t][v];
    } else {                                 // fp32 (token, dim) store
      float* o = (float*)outp;
#pragma unroll
      for (int t = 0; t < 4; ++t)
#pragma unroll
        for (int v = 0; v < 8; ++v) {
          int token = mb + v + half * 8;
          o[(size_t)token * DIM + n0 + t * 16 + col] = acc[mt][t][v];
        }
    }
  }
}

// ---------------------------------------------------------------------------
// Flash attention: 1 wave per (b, h, 16-query tile). 32 keys per iteration:
// 4 WMMAs for Q@K^T (K=64), online softmax in C-layout (shfl reductions
// within each 16-lane half), probs repacked C->A layout through 1KB LDS,
// 4 WMMAs for P@V. Causal mask applied analytically (== the mask input).
// ---------------------------------------------------------------------------
__global__ __launch_bounds__(32)
void attn_fa_k(const __bf16* __restrict__ qb, const __bf16* __restrict__ kb,
               const __bf16* __restrict__ vtb, __bf16* __restrict__ attnb) {
  __shared__ __bf16 sh[16 * 32];             // 16 queries x 32 keys probs tile

  const int lane = threadIdx.x & 31;
  const int half = lane >> 4;
  const int col  = lane & 15;
  const int q0   = blockIdx.x * 16;
  const int h    = blockIdx.y;
  const int b    = blockIdx.z;
  const int bh   = b * NH + h;

  const int oa1 = half * 8, oa2 = 16 + half * 8;     // A-operand chunks
  const int ob1 = half * 8, ob2 = 16 + half * 8;     // B-operand chunks

  // Q tile (A role): rows = queries, K = head_dim (two K=32 slabs)
  const __bf16* qrow = qb + (size_t)(bh * SEQ + q0 + col) * HD;
  v16bf qa0 = load_tile16(qrow, oa1, oa2);
  v16bf qa1 = load_tile16(qrow, 32 + oa1, 32 + oa2);

  const v8f vzero = {0.f, 0.f, 0.f, 0.f, 0.f, 0.f, 0.f, 0.f};
  v8f acc[4];
  float m_run[8], l_run[8];
#pragma unroll
  for (int t = 0; t < 4; ++t) acc[t] = vzero;
#pragma unroll
  for (int v = 0; v < 8; ++v) { m_run[v] = -3.0e38f; l_run[v] = 0.0f; }

  const int kend = q0 + 16;                  // causal: keys <= last query row
  for (int k0 = 0; k0 < kend; k0 += 32) {
    // ---- scores: two 16x16 tiles, K=64 accumulated over two WMMAs each
    v8f sc[2];
#pragma unroll
    for (int t = 0; t < 2; ++t) {
      const __bf16* krow = kb + (size_t)(bh * SEQ + k0 + t * 16 + col) * HD;
      v8f c = vzero;
      c = wmma_bf16(qa0, load_tile16(krow, ob1, ob2), c);
      c = wmma_bf16(qa1, load_tile16(krow, 32 + ob1, 32 + ob2), c);
      sc[t] = c;
    }
    // ---- online softmax (per C-layout row = v + half*8)
#pragma unroll
    for (int v = 0; v < 8; ++v) {
      int row = q0 + v + half * 8;
      float a0 = sc[0][v] * SM_SCALE + ((k0 + col)      > row ? NEG_INF : 0.0f);
      float a1 = sc[1][v] * SM_SCALE + ((k0 + 16 + col) > row ? NEG_INF : 0.0f);
      float mx = fmaxf(a0, a1);
#pragma unroll
      for (int off = 1; off < 16; off <<= 1) mx = fmaxf(mx, __shfl_xor(mx, off));
      float mn = fmaxf(m_run[v], mx);
      float p0 = __expf(a0 - mn);
      float p1 = __expf(a1 - mn);
      float sum = p0 + p1;
#pragma unroll
      for (int off = 1; off < 16; off <<= 1) sum += __shfl_xor(sum, off);
      float alpha = __expf(m_run[v] - mn);
      l_run[v] = l_run[v] * alpha + sum;
      m_run[v] = mn;
#pragma unroll
      for (int t = 0; t < 4; ++t) acc[t][v] *= alpha;
      sc[0][v] = p0; sc[1][v] = p1;
    }
    // ---- repack probs C-layout -> A-layout via LDS (16x32 bf16 tile)
#pragma unroll
    for (int t = 0; t < 2; ++t)
#pragma unroll
      for (int v = 0; v < 8; ++v)
        sh[(v + half * 8) * 32 + t * 16 + col] = (__bf16)sc[t][v];
    __syncthreads();
    v16bf pa = load_tile16(sh + col * 32, oa1, oa2);
    __syncthreads();
    // ---- P @ V : B = V^T tile, columns (= head dims) contiguous along s
#pragma unroll
    for (int t = 0; t < 4; ++t) {
      const __bf16* vrow = vtb + (size_t)(bh * HD + t * 16 + col) * SEQ + k0;
      acc[t] = wmma_bf16(pa, load_tile16(vrow, ob1, ob2), acc[t]);
    }
  }

  // ---- normalize and store bf16 (token, dim) for the output projection
#pragma unroll
  for (int v = 0; v < 8; ++v) {
    int row = q0 + v + half * 8;
    int token = b * SEQ + row;
    float inv = 1.0f / l_run[v];
#pragma unroll
    for (int t = 0; t < 4; ++t)
      attnb[(size_t)token * DIM + h * HD + t * 16 + col] =
          (__bf16)(acc[t][v] * inv);
  }
}

// ---------------------------------------------------------------------------
// Host-side orchestration.  Inputs: x, wq, wk, wv, wo, mask(unused: causal
// applied analytically), start_pos.  Workspace budget ~113 MiB.
// ---------------------------------------------------------------------------
extern "C" void kernel_launch(void* const* d_in, const int* in_sizes, int n_in,
                              void* d_out, int out_size, void* d_ws, size_t ws_size,
                              hipStream_t stream) {
  (void)in_sizes; (void)n_in; (void)out_size; (void)ws_size;
  const float* x  = (const float*)d_in[0];
  const float* wq = (const float*)d_in[1];
  const float* wk = (const float*)d_in[2];
  const float* wv = (const float*)d_in[3];
  const float* wo = (const float*)d_in[4];
  /* d_in[5] = mask: encoded analytically in the kernel */
  const int* start_pos = (const int*)d_in[6];

  char* ws = (char*)d_ws;
  size_t o = 0;
  auto carve = [&](size_t bytes) { char* p = ws + o; o += (bytes + 255) & ~(size_t)255; return p; };
  __bf16* xb    = (__bf16*)carve((size_t)TOK * DIM * 2);
  __bf16* wqb   = (__bf16*)carve((size_t)DIM * DIM * 2);
  __bf16* wkb   = (__bf16*)carve((size_t)DIM * DIM * 2);
  __bf16* wvb   = (__bf16*)carve((size_t)DIM * DIM * 2);
  __bf16* wob   = (__bf16*)carve((size_t)DIM * DIM * 2);
  __bf16* qb    = (__bf16*)carve((size_t)TOK * DIM * 2);   // (b,h,s,hd)
  __bf16* kb    = (__bf16*)carve((size_t)TOK * DIM * 2);   // (b,h,s,hd)
  __bf16* vtb   = (__bf16*)carve((size_t)TOK * DIM * 2);   // (b,h,hd,s)
  __bf16* attnb = (__bf16*)carve((size_t)TOK * DIM * 2);   // (token, dim)

  // 1) fp32 -> bf16 casts
  {
    int n4x = TOK * DIM / 4;
    cvt_f32_bf16_k<<<(n4x + 255) / 256, 256, 0, stream>>>(x, xb, n4x);
    int n4w = DIM * DIM / 4;
    cvt_f32_bf16_k<<<(n4w + 255) / 256, 256, 0, stream>>>(wq, wqb, n4w);
    cvt_f32_bf16_k<<<(n4w + 255) / 256, 256, 0, stream>>>(wk, wkb, n4w);
    cvt_f32_bf16_k<<<(n4w + 255) / 256, 256, 0, stream>>>(wv, wvb, n4w);
    cvt_f32_bf16_k<<<(n4w + 255) / 256, 256, 0, stream>>>(wo, wob, n4w);
  }

  // 2) projections (+RoPE for q,k; transpose for v)
  dim3 ggrid(DIM / 64, TOK / 256);
  gemm_bf16_k<0><<<ggrid, 256, 0, stream>>>(xb, wqb, (void*)qb,  start_pos);
  gemm_bf16_k<0><<<ggrid, 256, 0, stream>>>(xb, wkb, (void*)kb,  start_pos);
  gemm_bf16_k<1><<<ggrid, 256, 0, stream>>>(xb, wvb, (void*)vtb, start_pos);

  // 3) flash attention
  attn_fa_k<<<dim3(SEQ / 16, NH, BATCH), 32, 0, stream>>>(qb, kb, vtb, attnb);

  // 4) output projection -> fp32 d_out
  gemm_bf16_k<2><<<ggrid, 256, 0, stream>>>(attnb, wob, d_out, start_pos);
}
